// DKD_36060545417854
// MI455X (gfx1250) — compile-verified, matrix operand order checked by reading, not run on hardware
//
#include <hip/hip_runtime.h>
#include <hip/hip_fp16.h>
#include <stdint.h>

#define NB 16
#define IMH 1024
#define IMW 1024
#define TOPK 5000
#define SCORE_TH 0.2f
#define INV_TEMP 10.0f

#define TILE 64
#define S_DIM 72    /* TILE + 2*4 halo for fused double-NMS */
#define N1_DIM 68   /* TILE + 2*2 halo                      */
#define NGX 18      /* 72/4 float4 groups per staged row    */

#define CAND_CAP 65536
#define SEL_CAP 8192
#define HBINS 1024
#define PADK 5120

typedef unsigned long long u64;
typedef unsigned int u32;
typedef __attribute__((ext_vector_type(16))) _Float16 v16h;
typedef __attribute__((ext_vector_type(8))) float v8f;
typedef __attribute__((ext_vector_type(4))) int v4i;

#if defined(__gfx1250__) && __has_builtin(__builtin_amdgcn_global_load_async_to_lds_b128) && __has_builtin(__builtin_amdgcn_s_wait_asynccnt)
#define USE_ASYNC_LDS 1
typedef __attribute__((address_space(1))) v4i* gptr_v4i;
typedef __attribute__((address_space(3))) v4i* lptr_v4i;
#else
#define USE_ASYNC_LDS 0
#endif

__device__ __forceinline__ u64 pack_key(float s, u32 idx) {
  return ((u64)__float_as_uint(s) << 32) | (u32)(~idx);
}
__device__ __forceinline__ int score_bin(float s) {
  int bin = (int)((s - SCORE_TH) * ((float)HBINS / 0.8f));
  return min(max(bin, 0), HBINS - 1);
}

__global__ void dkd_init_kernel(int* cand_cnt, int* sel_cnt, int* hist) {
  int i = threadIdx.x;
  if (i < NB) { cand_cnt[i] = 0; sel_cnt[i] = 0; }
  for (int j = i; j < NB * HBINS; j += 1024) hist[j] = 0;
}

// Fused double-NMS over 64x64 tiles with separable 5x5 window max;
// emits (score,idx) survivors > TH directly (NMS map never materialized).
__global__ __launch_bounds__(256) void dkd_nms_kernel(
    const float* __restrict__ sm, u64* __restrict__ cand, int* __restrict__ cand_cnt) {
  __shared__ float s_s[S_DIM * S_DIM];       // staged scores, halo 4
  __shared__ float s_rm[S_DIM * N1_DIM];     // row-max plane (reused for pass 2)
  __shared__ float s_n1[N1_DIM * N1_DIM];    // nms1, halo 2
  const int b = blockIdx.y;
  const int oy = (blockIdx.x >> 4) * TILE;
  const int ox = (blockIdx.x & 15) * TILE;
  const float* img = sm + (size_t)b * IMH * IMW;
  const int tid = threadIdx.x;
  const float NEG_INF = -__builtin_inff();

  // Stage score tile (halo 4) into LDS in float4 groups; async-to-LDS on CDNA5.
  // ox-4 is a multiple of 4 and IMW%4==0, so each group is fully in or fully out.
  for (int g = tid; g < S_DIM * NGX; g += 256) {
    int ly = g / NGX, gi = g - ly * NGX;
    int gy = oy - 4 + ly;
    int gx = ox - 4 + gi * 4;
    int li = ly * S_DIM + gi * 4;
    bool inb = (gy >= 0) & (gy < IMH) & (gx >= 0) & (gx + 3 < IMW);
    if (inb) {
#if USE_ASYNC_LDS
      __builtin_amdgcn_global_load_async_to_lds_b128(
          (gptr_v4i)(img + (size_t)gy * IMW + gx), (lptr_v4i)&s_s[li], 0, 0);
#else
      *(float4*)&s_s[li] = *(const float4*)(img + (size_t)gy * IMW + gx);
#endif
    } else {
      s_s[li + 0] = NEG_INF; s_s[li + 1] = NEG_INF;
      s_s[li + 2] = NEG_INF; s_s[li + 3] = NEG_INF;
    }
  }
#if USE_ASYNC_LDS
  __builtin_amdgcn_s_wait_asynccnt(0);
#endif
  __syncthreads();

  // Pass 1a: row max (x +/- 2) over all 72 rows, 68 output columns.
  for (int i = tid; i < S_DIM * N1_DIM; i += 256) {
    int y = i / N1_DIM, x = i - y * N1_DIM;
    const float* r = &s_s[y * S_DIM + x];
    float m = fmaxf(fmaxf(fmaxf(r[0], r[1]), fmaxf(r[2], r[3])), r[4]);
    s_rm[i] = m;
  }
  __syncthreads();

  // Pass 1b: column max of row-max => mp1; nms1 = (s == mp1) ? s : 0 on 68x68.
  for (int i = tid; i < N1_DIM * N1_DIM; i += 256) {
    int ly = i / N1_DIM, lx = i - ly * N1_DIM;
    int sy = ly + 2;
    float v = s_s[sy * S_DIM + lx + 2];
    const float* c = &s_rm[ly * N1_DIM + lx];   // rows sy-2 .. sy+2
    float m = fmaxf(fmaxf(fmaxf(c[0], c[N1_DIM]), fmaxf(c[2 * N1_DIM], c[3 * N1_DIM])),
                    c[4 * N1_DIM]);
    s_n1[i] = (v == m) ? v : 0.0f;
  }
  __syncthreads();

  // Pass 2a: row max of nms1 (68 rows x 64 cols), reusing s_rm.
  for (int i = tid; i < N1_DIM * TILE; i += 256) {
    int y = i / TILE, x = i - y * TILE;
    const float* r = &s_n1[y * N1_DIM + x];
    s_rm[i] = fmaxf(fmaxf(fmaxf(r[0], r[1]), fmaxf(r[2], r[3])), r[4]);
  }
  __syncthreads();

  // Pass 2b: column max + survivor emission on the 64x64 interior.
  for (int i = tid; i < TILE * TILE; i += 256) {
    int ly = i >> 6, lx = i & 63;
    float v = s_n1[(ly + 2) * N1_DIM + lx + 2];
    if (v > SCORE_TH) {
      const float* c = &s_rm[ly * TILE + lx];   // nms1 rows ly .. ly+4
      float m = fmaxf(fmaxf(fmaxf(c[0], c[TILE]), fmaxf(c[2 * TILE], c[3 * TILE])),
                      c[4 * TILE]);
      if (v == m) {
        u32 idx = (u32)((oy + ly) * IMW + (ox + lx));
        int pos = atomicAdd(&cand_cnt[b], 1);
        if (pos < CAND_CAP) cand[(size_t)b * CAND_CAP + pos] = pack_key(v, idx);
      }
    }
  }
}

__global__ void dkd_hist_kernel(const u64* __restrict__ cand, const int* __restrict__ cand_cnt,
                                int* __restrict__ hist) {
  int b = blockIdx.y;
  int n = min(cand_cnt[b], CAND_CAP);
  for (int i = blockIdx.x * blockDim.x + threadIdx.x; i < n; i += gridDim.x * blockDim.x) {
    float s = __uint_as_float((u32)(cand[(size_t)b * CAND_CAP + i] >> 32));
    atomicAdd(&hist[b * HBINS + score_bin(s)], 1);
  }
}

// Find cutoff bin (cumulative-from-top >= TOPK), select candidates >= cutoff bin.
__global__ __launch_bounds__(1024) void dkd_select_kernel(
    const u64* __restrict__ cand, const int* __restrict__ cand_cnt,
    const int* __restrict__ hist, u64* __restrict__ sel, int* __restrict__ sel_cnt) {
  int b = blockIdx.x;
  __shared__ int s_cut;
  if (threadIdx.x == 0) {
    int acc = 0, cut = 0;
    for (int bin = HBINS - 1; bin >= 0; --bin) {
      acc += hist[b * HBINS + bin];
      if (acc >= TOPK) { cut = bin; break; }
    }
    s_cut = cut;
  }
  __syncthreads();
  int cut = s_cut;
  int n = min(cand_cnt[b], CAND_CAP);
  for (int i = threadIdx.x; i < n; i += 1024) {
    u64 key = cand[(size_t)b * CAND_CAP + i];
    float s = __uint_as_float((u32)(key >> 32));
    if (score_bin(s) >= cut) {
      int pos = atomicAdd(&sel_cnt[b], 1);
      if (pos < SEL_CAP) sel[(size_t)b * SEL_CAP + pos] = key;
    }
  }
}

// Bitonic sort 8192 u64 keys (descending) in 64KB LDS; one workgroup per image.
// Key = (score_bits<<32) | ~idx: one u64 compare gives JAX's tie-break order.
__global__ __launch_bounds__(1024) void dkd_sort_kernel(
    const u64* __restrict__ sel, const int* __restrict__ sel_cnt,
    float* __restrict__ out_scores, int* __restrict__ top_idx) {
  __shared__ u64 skeys[SEL_CAP];
  int b = blockIdx.x;
  int n = min(sel_cnt[b], SEL_CAP);
  for (int i = threadIdx.x; i < SEL_CAP; i += 1024)
    skeys[i] = (i < n) ? sel[(size_t)b * SEL_CAP + i] : 0ull;
  __syncthreads();
  for (int k = 2; k <= SEL_CAP; k <<= 1) {
    for (int j = k >> 1; j > 0; j >>= 1) {
      for (int i = threadIdx.x; i < SEL_CAP; i += 1024) {
        int ixj = i ^ j;
        if (ixj > i) {
          u64 a = skeys[i], c = skeys[ixj];
          bool up = ((i & k) == 0);
          if (up ? (a < c) : (a > c)) { skeys[i] = c; skeys[ixj] = a; }
        }
      }
      __syncthreads();
    }
  }
  for (int j = threadIdx.x; j < TOPK; j += 1024) {
    u64 key = skeys[j];
    float s = __uint_as_float((u32)(key >> 32));
    out_scores[b * TOPK + j] = s;  // padding keys give 0.0 (matches reference zeroing)
    top_idx[b * PADK + j] = (s > SCORE_TH) ? (int)(~(u32)key) : -1;
  }
}

// Softmax-weighted sub-pixel refinement; offsets via v_wmma_f32_16x16x32_f16.
// A (16x32 f16): 16 keypoints x 25 softmax weights (zero pad); B col0=dx, col1=dy.
__global__ __launch_bounds__(256) void dkd_refine_kernel(
    const float* __restrict__ sm, const float* __restrict__ top_scores,
    const int* __restrict__ top_idx, float* __restrict__ out_kpts) {
  __shared__ float s_offx[8][16];
  __shared__ float s_offy[8][16];
  const int b = blockIdx.y;
  const int wave = threadIdx.x >> 5;
  const int lane = threadIdx.x & 31;
  const int N = lane & 15;   // keypoint row (A) / output column (B,D)
  const int hf = lane >> 4;  // which K-half this lane holds
  const int kp = (blockIdx.x * 8 + wave) * 16 + N;

  int j = (kp < TOPK) ? kp : (TOPK - 1);
  float score = top_scores[b * TOPK + j];
  int idx = top_idx[b * PADK + j];
  bool valid = (kp < TOPK) && (score > SCORE_TH) && (idx >= 0);
  int yc = valid ? (idx >> 10) : 512;
  int xc = valid ? (idx & (IMW - 1)) : 512;
  const float* img = sm + (size_t)b * IMH * IMW;

  float logit[25];
  float maxv = -__builtin_inff();
#pragma unroll
  for (int t = 0; t < 25; ++t) {
    int yy = yc + t / 5 - 2;
    int xx = xc + t % 5 - 2;
    bool inb = (yy >= 0) & (yy < IMH) & (xx >= 0) & (xx < IMW);
    float l = inb ? img[(size_t)yy * IMW + xx] * INV_TEMP : -1.0e9f;
    logit[t] = l;
    maxv = fmaxf(maxv, l);
  }
  float sum = 0.0f, w[25];
#pragma unroll
  for (int t = 0; t < 25; ++t) { w[t] = __expf(logit[t] - maxv); sum += w[t]; }
  float inv = 1.0f / sum;

  // CDNA5 16-bit A layout: lane m holds K 0-7,16-23; lane m+16 holds K 8-15,24-31.
  v16h a, bb;
#pragma unroll
  for (int e = 0; e < 16; ++e) {
    int K = (e < 8 ? e : e + 8) + hf * 8;
    a[e] = (_Float16)((K < 25) ? w[K] * inv : 0.0f);
    float bv = 0.0f;
    if (K < 25) {
      if (N == 0) bv = (float)(K % 5 - 2);       // dx
      else if (N == 1) bv = (float)(K / 5 - 2);  // dy
    }
    bb[e] = (_Float16)bv;
  }
  v8f c = {};
  c = __builtin_amdgcn_wmma_f32_16x16x32_f16(false, a, false, bb, (short)0, c, false, false);

  // D layout: lane n (+16) VGPR r = D[M=r(+8)][N=n]. Columns 0/1 carry offsets.
  if (N == 0) {
#pragma unroll
    for (int r = 0; r < 8; ++r) s_offx[wave][hf * 8 + r] = c[r];
  } else if (N == 1) {
#pragma unroll
    for (int r = 0; r < 8; ++r) s_offy[wave][hf * 8 + r] = c[r];
  }
  __syncthreads();

  if (hf == 0 && kp < TOPK) {
    float ox = s_offx[wave][N], oy = s_offy[wave][N];
    out_kpts[((size_t)b * TOPK + kp) * 2 + 0] = valid ? (float)xc + ox : 0.0f;
    out_kpts[((size_t)b * TOPK + kp) * 2 + 1] = valid ? (float)yc + oy : 0.0f;
  }
}

extern "C" void kernel_launch(void* const* d_in, const int* in_sizes, int n_in,
                              void* d_out, int out_size, void* d_ws, size_t ws_size,
                              hipStream_t stream) {
  (void)in_sizes; (void)n_in; (void)out_size; (void)ws_size;
  const float* sm = (const float*)d_in[0];
  float* out_kpts = (float*)d_out;                  // 16*5000*2
  float* out_scores = out_kpts + NB * TOPK * 2;     // 16*5000

  char* p = (char*)d_ws;
  u64* cand = (u64*)p;      p += (size_t)NB * CAND_CAP * sizeof(u64);
  u64* sel = (u64*)p;       p += (size_t)NB * SEL_CAP * sizeof(u64);
  int* cand_cnt = (int*)p;  p += NB * sizeof(int);
  int* sel_cnt = (int*)p;   p += NB * sizeof(int);
  int* hist = (int*)p;      p += NB * HBINS * sizeof(int);
  int* top_idx = (int*)p;   p += NB * PADK * sizeof(int);

  dkd_init_kernel<<<1, 1024, 0, stream>>>(cand_cnt, sel_cnt, hist);
  dkd_nms_kernel<<<dim3(256, NB), 256, 0, stream>>>(sm, cand, cand_cnt);
  dkd_hist_kernel<<<dim3(32, NB), 256, 0, stream>>>(cand, cand_cnt, hist);
  dkd_select_kernel<<<NB, 1024, 0, stream>>>(cand, cand_cnt, hist, sel, sel_cnt);
  dkd_sort_kernel<<<NB, 1024, 0, stream>>>(sel, sel_cnt, out_scores, top_idx);
  dkd_refine_kernel<<<dim3(40, NB), 256, 0, stream>>>(sm, out_scores, top_idx, out_kpts);
}
